// NeighborhoodFusionTransformer_37323265802415
// MI455X (gfx1250) — matrix operand order repaired and checked by reading, not compiled, not run on hardware
//
#include <hip/hip_runtime.h>

namespace {

constexpr int kD   = 256;
constexpr int kNH  = 8;
constexpr int kHD  = 32;
constexpr int kB   = 4;
constexpr int kN   = 784;            // tokens per image
constexpr int kR   = kB * kN;        // 3136 flattened rows
constexpr int kVTLD = 800;           // padded key stride for V^T rows (32B aligned: 800*2=1600)

typedef _Float16 h16;
typedef __attribute__((ext_vector_type(16))) _Float16 v16h;
typedef __attribute__((ext_vector_type(8)))  _Float16 v8h;
typedef __attribute__((ext_vector_type(8)))  float    v8f;

union V16 { v16h v; v8h h[2]; };

__device__ __forceinline__ v8f wmma_f32_f16(v16h a, v16h b, v8f c) {
  // D = A(16x32 f16) x B(32x16 f16) + C(16x16 f32)
  return __builtin_amdgcn_wmma_f32_16x16x32_f16(false, a, false, b, (short)0, c, false, false);
}

__device__ __forceinline__ float gelu_exact(float x) {
  return 0.5f * x * (1.0f + erff(x * 0.7071067811865476f));
}

// ---------------------------------------------------------------- converts
__global__ __launch_bounds__(256) void k_f32_to_f16(const float* __restrict__ in,
                                                    h16* __restrict__ out, int n) {
  int i = blockIdx.x * 256 + threadIdx.x;
  if (i < n) out[i] = (h16)in[i];
}

// f[B,D,28,28] (NCHW) -> s[R, D] fp32 + f16 mirror
__global__ __launch_bounds__(256) void k_to_seq(const float* __restrict__ f,
                                                float* __restrict__ s32,
                                                h16* __restrict__ s16) {
  int i = blockIdx.x * 256 + threadIdx.x;  // exactly kR*kD threads
  int d = i & (kD - 1);
  int n = i >> 8;
  int b = n / kN, pp = n % kN;
  float v = f[(b * kD + d) * kN + pp];
  s32[i] = v;
  s16[i] = (h16)v;
}

__global__ __launch_bounds__(256) void k_from_seq(const float* __restrict__ x,
                                                  float* __restrict__ out) {
  int i = blockIdx.x * 256 + threadIdx.x;  // exactly kR*kD threads
  int d = i & (kD - 1);
  int n = i >> 8;
  int b = n / kN, pp = n % kN;
  out[(b * kD + d) * kN + pp] = x[i];
}

__global__ __launch_bounds__(256) void k_avg(const float* __restrict__ a,
                                             const float* __restrict__ b,
                                             float* __restrict__ o, int n) {
  int i = blockIdx.x * 256 + threadIdx.x;
  if (i < n) o[i] = 0.5f * (a[i] + b[i]);
}

// v[R, ldv] head-sliced -> vt[(b*NH+h)][d=32][key padded to 800], zero padded
__global__ __launch_bounds__(256) void k_vtrans(const h16* __restrict__ v, int ldv,
                                                h16* __restrict__ vt) {
  const int total = kB * kNH * kHD * kVTLD;
  int i = blockIdx.x * 256 + threadIdx.x;
  if (i >= total) return;
  const int key  = i % kVTLD;
  const int rest = i / kVTLD;
  const int d  = rest & (kHD - 1);
  const int bh = rest >> 5;
  const int b = bh >> 3;
  const int h = bh & 7;
  h16 val = (h16)0.0f;
  if (key < kN) val = v[(b * kN + key) * ldv + h * kHD + d];
  vt[i] = val;
}

// ---------------------------------------------------------------- WMMA GEMM
// Y[row, col] = act( sum_k A[row,k] * W[k,col] + bias[col] )
// A: [kR x K] f16 (lda row stride), W: [K x N] f16 row-major.
// Tile: BM=64 x BN=128, BK=32, double-buffered LDS.
// 8 waves: 4 in M x 2 in N; each wave computes 16x64 = 4 WMMA tiles per K-step.
// Requires kR%64==0, K%32==0, N%128==0 (true for all call sites).
__global__ __launch_bounds__(256) void k_gemm(
    const h16* __restrict__ A, int lda,
    const h16* __restrict__ W,
    const float* __restrict__ bias,
    float* __restrict__ outF, h16* __restrict__ outH, int ldo,
    int K, int N, int act) {
  __shared__ h16 As[2][64 * 32];     // [row][k]
  __shared__ h16 Bst[2][128 * 32];   // transposed: [col][k]

  const int tid  = threadIdx.x;
  const int wid  = tid >> 5;
  const int lane = tid & 31;
  const int l16  = lane & 15;
  const int hi   = lane >> 4;           // 0: lanes 0-15, 1: lanes 16-31
  const int rowBase = blockIdx.x * 64;
  const int colBase = blockIdx.y * 128;
  const int m_off = (wid & 3) * 16;
  const int n_off = (wid >> 2) * 64;

  const int arow = tid >> 2, aseg = tid & 3;   // A stage: 64 rows x 4 segs of 8
  const int brow = tid >> 3, bseg = tid & 7;   // W stage: 32 k-rows x 8 segs of 16 cols

  const h16* aSrc = A + (rowBase + arow) * lda + aseg * 8;     // + k0
  const h16* bSrc = W + brow * N + colBase + bseg * 16;        // + k0*N

  v8h aReg, bReg0, bReg1;
  auto gload = [&](int k0) {
    aReg  = *(const v8h*)(aSrc + k0);
    bReg0 = *(const v8h*)(bSrc + k0 * N);
    bReg1 = *(const v8h*)(bSrc + k0 * N + 8);
  };
  auto lstore = [&](int buf) {
    *(v8h*)&As[buf][arow * 32 + aseg * 8] = aReg;
    h16* bp = &Bst[buf][0];
#pragma unroll
    for (int j = 0; j < 8; ++j) bp[(bseg * 16 + j) * 32 + brow] = bReg0[j];
#pragma unroll
    for (int j = 0; j < 8; ++j) bp[(bseg * 16 + 8 + j) * 32 + brow] = bReg1[j];
  };

  v8f acc[4] = {};

  gload(0);
  lstore(0);

  for (int k0 = 0, buf = 0; k0 < K; k0 += 32, buf ^= 1) {
    __syncthreads();                       // buf fully staged
    const bool more = (k0 + 32) < K;
    if (more) {
      gload(k0 + 32);                      // in flight during compute
      if (k0 + 64 < K) {                   // warm L2/WGP$ two steps ahead
        __builtin_prefetch(aSrc + k0 + 64, 0, 1);
        __builtin_prefetch(bSrc + (k0 + 64) * N, 0, 1);
      }
    }

    // Load ALL fragments into distinct registers first, then fire the WMMA
    // burst back-to-back (single dscnt wait instead of one per WMMA).
    // A fragment (16x32 f16): lane<16 holds K 0..15, lane>=16 K 16..31
    V16 af;
    const int ao = (m_off + l16) * 32 + hi * 16;
    af.h[0] = *(const v8h*)&As[buf][ao];
    af.h[1] = *(const v8h*)&As[buf][ao + 8];

    // B fragments (32x16 f16): lane<16 holds K 0..15, lane>=16 K 16..31, col=l16
    V16 bf[4];
#pragma unroll
    for (int t = 0; t < 4; ++t) {
      const int bo = (n_off + t * 16 + l16) * 32 + hi * 16;
      bf[t].h[0] = *(const v8h*)&Bst[buf][bo];
      bf[t].h[1] = *(const v8h*)&Bst[buf][bo + 8];
    }
#pragma unroll
    for (int t = 0; t < 4; ++t)
      acc[t] = wmma_f32_f16(af.v, bf[t].v, acc[t]);

    __syncthreads();                       // all reads of buf done
    if (more) lstore(buf ^ 1);
  }

#pragma unroll
  for (int t = 0; t < 4; ++t) {
    const int col = colBase + n_off + t * 16 + l16;
    const float bv = bias ? bias[col] : 0.0f;
#pragma unroll
    for (int r = 0; r < 8; ++r) {
      const int row = rowBase + m_off + r + hi * 8;   // C layout: M = r + 8*hi
      float v = acc[t][r] + bv;
      if (act == 1) v = gelu_exact(v);
      const int idx = row * ldo + col;
      if (outF) outF[idx] = v;
      if (outH) outH[idx] = (h16)v;
    }
  }
}

// ---------------------------------------------------------------- flash attention
// One wave = 16 query rows of one (b,h). Online softmax over 32-key chunks.
// Q:[b*784+q, ldq] f16 (head slice at h*32), K same, Vt:[(b*8+h)][32][kVTLD] f16.
// Out f16 at [b*784+q, ldo], col h*32 + d  (direction offset baked into Out ptr).
__global__ __launch_bounds__(256) void k_attn(
    const h16* __restrict__ Q, int ldq,
    const h16* __restrict__ Km, int ldk,
    const h16* __restrict__ Vt,
    h16* __restrict__ Out, int ldo,
    float scale) {
  __shared__ h16 Plds[8 * 16 * 32];  // per-wave 16x32 P tile

  const int tid  = threadIdx.x;
  const int wid  = tid >> 5;
  const int lane = tid & 31;
  const int l16  = lane & 15;
  const int hi   = lane >> 4;
  const int bh = blockIdx.y;
  const int b  = bh >> 3;
  const int h  = bh & 7;
  const int hq = h * kHD;
  const int qbase = blockIdx.x * 128 + wid * 16;

  int qr = qbase + l16;
  if (qr >= kN) qr = kN - 1;                    // clamp loads; stores masked later
  const h16* qp = &Q[(b * kN + qr) * ldq + hq + hi * 16];
  V16 aq;                                       // Q as A fragment (16x32, K=d)
  aq.h[0] = *(const v8h*)qp;
  aq.h[1] = *(const v8h*)(qp + 8);

  v8f o0 = {}, o1 = {};                         // O accumulators, d cols 0..15 / 16..31
  float mi[8], li[8];
#pragma unroll
  for (int r = 0; r < 8; ++r) { mi[r] = -1e30f; li[r] = 0.0f; }

  h16* P = &Plds[wid * (16 * 32)];
  const h16* vtb = &Vt[bh * kHD * kVTLD];

  for (int j0 = 0; j0 < kN; j0 += 32) {
    // K^T fragments: B[d][key] = K[key][d] -> contiguous d loads per key row
    int k0r = j0 + l16;      if (k0r >= kN) k0r = kN - 1;
    int k1r = j0 + 16 + l16; if (k1r >= kN) k1r = kN - 1;
    const h16* kp0 = &Km[(b * kN + k0r) * ldk + hq + hi * 16];
    const h16* kp1 = &Km[(b * kN + k1r) * ldk + hq + hi * 16];
    V16 bk0, bk1;
    bk0.h[0] = *(const v8h*)kp0; bk0.h[1] = *(const v8h*)(kp0 + 8);
    bk1.h[0] = *(const v8h*)kp1; bk1.h[1] = *(const v8h*)(kp1 + 8);

    v8f s0 = {}, s1 = {};
    s0 = wmma_f32_f16(aq.v, bk0.v, s0);
    s1 = wmma_f32_f16(aq.v, bk1.v, s1);

    const bool v0 = (j0 + l16) < kN;
    const bool v1 = (j0 + 16 + l16) < kN;
    float mx[8];
#pragma unroll
    for (int r = 0; r < 8; ++r) {
      float a0 = v0 ? s0[r] * scale : -1e30f;
      float a1 = v1 ? s1[r] * scale : -1e30f;
      s0[r] = a0; s1[r] = a1;
      mx[r] = fmaxf(a0, a1);
    }
    // row reductions across the 16 lanes of each half-group (xor 1,2,4,8 stays in-half)
#pragma unroll
    for (int m = 1; m <= 8; m <<= 1)
#pragma unroll
      for (int r = 0; r < 8; ++r) mx[r] = fmaxf(mx[r], __shfl_xor(mx[r], m, 32));

    float p0[8], p1[8], rs[8], al[8];
#pragma unroll
    for (int r = 0; r < 8; ++r) {
      float mn = fmaxf(mi[r], mx[r]);
      al[r] = __expf(mi[r] - mn);
      mi[r] = mn;
      p0[r] = __expf(s0[r] - mn);
      p1[r] = __expf(s1[r] - mn);
      rs[r] = p0[r] + p1[r];
    }
#pragma unroll
    for (int m = 1; m <= 8; m <<= 1)
#pragma unroll
      for (int r = 0; r < 8; ++r) rs[r] += __shfl_xor(rs[r], m, 32);
#pragma unroll
    for (int r = 0; r < 8; ++r) {
      li[r] = li[r] * al[r] + rs[r];
      o0[r] *= al[r];
      o1[r] *= al[r];
    }

    // P: C layout -> LDS -> reload as A fragment (16 rows x 32 keys)
#pragma unroll
    for (int r = 0; r < 8; ++r) {
      const int prow = r + hi * 8;
      P[prow * 32 + l16]      = (h16)p0[r];
      P[prow * 32 + 16 + l16] = (h16)p1[r];
    }
    asm volatile("s_wait_dscnt 0" ::: "memory");  // wave-internal LDS RAW fence

    // Load P fragment + both V fragments before the PV WMMA pair.
    V16 ap;
    const int po = l16 * 32 + hi * 16;
    ap.h[0] = *(const v8h*)&P[po];
    ap.h[1] = *(const v8h*)&P[po + 8];

    // V fragments: B[key][d] from Vt[d][key] -> contiguous key loads
    const h16* vp0 = &vtb[l16 * kVTLD + j0 + hi * 16];
    const h16* vp1 = &vtb[(16 + l16) * kVTLD + j0 + hi * 16];
    V16 bv0, bv1;
    bv0.h[0] = *(const v8h*)vp0; bv0.h[1] = *(const v8h*)(vp0 + 8);
    bv1.h[0] = *(const v8h*)vp1; bv1.h[1] = *(const v8h*)(vp1 + 8);

    o0 = wmma_f32_f16(ap.v, bv0.v, o0);
    o1 = wmma_f32_f16(ap.v, bv1.v, o1);
  }

#pragma unroll
  for (int r = 0; r < 8; ++r) {
    const int q = qbase + r + hi * 8;
    if (q < kN) {
      const float inv = 1.0f / li[r];
      Out[(b * kN + q) * ldo + hq + l16]      = (h16)(o0[r] * inv);
      Out[(b * kN + q) * ldo + hq + 16 + l16] = (h16)(o1[r] * inv);
    }
  }
}

// ---------------------------------------------------------------- layernorm (+gated residual)
// out = LN( X + tanh(gate)*Y ; g, b ), one row per block, C in {256, 1024}
__global__ __launch_bounds__(256) void k_ln(
    const float* __restrict__ X, const float* __restrict__ Y,
    const float* __restrict__ gateptr,
    const float* __restrict__ lng, const float* __restrict__ lnb,
    float* __restrict__ outF, h16* __restrict__ outH, int C) {
  __shared__ float red[256];
  const int row = blockIdx.x;
  const int tid = threadIdx.x;
  const int per = C >> 8;
  float coef = 1.0f;
  if (gateptr) coef = tanhf(gateptr[0]);
  const float* xr = X + row * C;
  const float* yr = Y ? Y + row * C : nullptr;
  float vals[4];
  float s = 0.0f;
  for (int i = 0; i < per; ++i) {
    const int c = tid + i * 256;
    float v = xr[c];
    if (yr) v += coef * yr[c];
    vals[i] = v;
    s += v;
  }
  red[tid] = s; __syncthreads();
  for (int st = 128; st > 0; st >>= 1) { if (tid < st) red[tid] += red[tid + st]; __syncthreads(); }
  const float mean = red[0] / (float)C;
  __syncthreads();
  float s2 = 0.0f;
  for (int i = 0; i < per; ++i) { float d = vals[i] - mean; s2 += d * d; }
  red[tid] = s2; __syncthreads();
  for (int st = 128; st > 0; st >>= 1) { if (tid < st) red[tid] += red[tid + st]; __syncthreads(); }
  const float rstd = rsqrtf(red[0] / (float)C + 1e-5f);
  for (int i = 0; i < per; ++i) {
    const int c = tid + i * 256;
    float o = (vals[i] - mean) * rstd * lng[c] + lnb[c];
    if (outF) outF[row * C + c] = o;
    if (outH) outH[row * C + c] = (h16)o;
  }
}

}  // namespace

// ================================================================ host
extern "C" void kernel_launch(void* const* d_in, const int* in_sizes, int n_in,
                              void* d_out, int out_size, void* d_ws, size_t ws_size,
                              hipStream_t stream) {
  const float* f[4] = {(const float*)d_in[0], (const float*)d_in[1],
                       (const float*)d_in[2], (const float*)d_in[3]};
  const float* Wq_a = (const float*)d_in[4];
  const float* Wk_a = (const float*)d_in[5];
  const float* Wv_a = (const float*)d_in[6];
  const float* Wq_b = (const float*)d_in[7];
  const float* Wk_b = (const float*)d_in[8];
  const float* Wv_b = (const float*)d_in[9];
  const float* bq_a = (const float*)d_in[10];
  const float* bk_a = (const float*)d_in[11];
  const float* bv_a = (const float*)d_in[12];
  const float* bq_b = (const float*)d_in[13];
  const float* bk_b = (const float*)d_in[14];
  const float* bv_b = (const float*)d_in[15];
  const float* gamma     = (const float*)d_in[16];
  const float* gamma_ffn = (const float*)d_in[17];
  const float* Wfuse = (const float*)d_in[18];
  const float* bfuse = (const float*)d_in[19];
  const float* W1 = (const float*)d_in[20];
  const float* b1 = (const float*)d_in[21];
  const float* W2 = (const float*)d_in[22];
  const float* b2 = (const float*)d_in[23];
  const float* ln1_g = (const float*)d_in[24];
  const float* ln1_b = (const float*)d_in[25];
  const float* ln2_g = (const float*)d_in[26];
  const float* ln2_b = (const float*)d_in[27];
  const float* fp_ln_g = (const float*)d_in[28];
  const float* fp_ln_b = (const float*)d_in[29];
  const float* fp_W = (const float*)d_in[30];
  const float* fp_b = (const float*)d_in[31];
  const float* ff_ln_g = (const float*)d_in[32];
  const float* ff_ln_b = (const float*)d_in[33];
  const float* ff_W = (const float*)d_in[34];
  const float* ff_b = (const float*)d_in[35];
  const float* mha_Wqkv = (const float*)d_in[36];
  const float* mha_bqkv = (const float*)d_in[37];
  const float* mha_Wo = (const float*)d_in[38];
  const float* mha_bo = (const float*)d_in[39];
  const float* norm_g = (const float*)d_in[40];
  const float* norm_b = (const float*)d_in[41];

  if (ws_size < (size_t)96 * 1024 * 1024) return;  // needs ~82 MB scratch

  char* p = (char*)d_ws;
  auto alloc = [&](size_t bytes) -> void* {
    void* r = (void*)p;
    p += (bytes + 255) & ~(size_t)255;
    return r;
  };

  const int RD = kR * kD;  // 802816

  // ---- persistent arena: f16 weights, sequences, block-average outputs
  h16* wq_a16 = (h16*)alloc((size_t)4 * kD * kD * 2);
  h16* wk_a16 = (h16*)alloc((size_t)4 * kD * kD * 2);
  h16* wv_a16 = (h16*)alloc((size_t)4 * kD * kD * 2);
  h16* wq_b16 = (h16*)alloc((size_t)4 * kD * kD * 2);
  h16* wk_b16 = (h16*)alloc((size_t)4 * kD * kD * 2);
  h16* wv_b16 = (h16*)alloc((size_t)4 * kD * kD * 2);
  h16* wfuse16 = (h16*)alloc((size_t)4 * 2 * kD * kD * 2);
  h16* w1_16 = (h16*)alloc((size_t)4 * kD * 4 * kD * 2);
  h16* w2_16 = (h16*)alloc((size_t)4 * 4 * kD * kD * 2);
  h16* fpW16 = (h16*)alloc((size_t)kD * kD * 2);
  h16* ffW16 = (h16*)alloc((size_t)4 * kD * kD * 2);
  h16* wqkv16 = (h16*)alloc((size_t)kD * 3 * kD * 2);
  h16* wo16 = (h16*)alloc((size_t)kD * kD * 2);
  float* s32[4]; h16* s16[4]; float* e32[4];
  for (int i = 0; i < 4; ++i) s32[i] = (float*)alloc((size_t)RD * 4);
  for (int i = 0; i < 4; ++i) s16[i] = (h16*)alloc((size_t)RD * 2);
  for (int i = 0; i < 4; ++i) e32[i] = (float*)alloc((size_t)RD * 4);

  char* transBase = p;

  // ---- helpers
  auto conv = [&](const float* src, h16* dst, int n) {
    k_f32_to_f16<<<(n + 255) / 256, 256, 0, stream>>>(src, dst, n);
  };
  auto gemm = [&](const h16* A, int lda, const h16* W, const float* bias,
                  float* oF, h16* oH, int ldo, int K, int N, int act) {
    dim3 g(kR / 64, N / 128);
    k_gemm<<<g, 256, 0, stream>>>(A, lda, W, bias, oF, oH, ldo, K, N, act);
  };
  auto attn = [&](const h16* Q, int ldq, const h16* Km, int ldk, const h16* Vt,
                  h16* Out, int ldo) {
    dim3 g((kN + 127) / 128, kB * kNH);
    k_attn<<<g, 256, 0, stream>>>(Q, ldq, Km, ldk, Vt, Out, ldo, 0.17677669529663689f);
  };
  auto ln = [&](const float* X, const float* Y, const float* gate,
                const float* g_, const float* b_, float* oF, h16* oH, int C) {
    k_ln<<<kR, 256, 0, stream>>>(X, Y, gate, g_, b_, oF, oH, C);
  };
  auto vtrans = [&](const h16* v, int ldv, h16* vt) {
    int n = kB * kNH * kHD * kVTLD;
    k_vtrans<<<(n + 255) / 256, 256, 0, stream>>>(v, ldv, vt);
  };

  // ---- weight conversion + sequence layout
  conv(Wq_a, wq_a16, 4 * kD * kD);  conv(Wk_a, wk_a16, 4 * kD * kD);
  conv(Wv_a, wv_a16, 4 * kD * kD);  conv(Wq_b, wq_b16, 4 * kD * kD);
  conv(Wk_b, wk_b16, 4 * kD * kD);  conv(Wv_b, wv_b16, 4 * kD * kD);
  conv(Wfuse, wfuse16, 4 * 2 * kD * kD);
  conv(W1, w1_16, 4 * kD * 4 * kD);
  conv(W2, w2_16, 4 * 4 * kD * kD);
  conv(fp_W, fpW16, kD * kD);
  conv(ff_W, ffW16, 4 * kD * kD);
  conv(mha_Wqkv, wqkv16, kD * 3 * kD);
  conv(mha_Wo, wo16, kD * kD);
  for (int i = 0; i < 4; ++i)
    k_to_seq<<<RD / 256, 256, 0, stream>>>(f[i], s32[i], s16[i]);

  // ---- block-phase transient arena
  h16* qa = (h16*)alloc((size_t)RD * 2);
  h16* kbuf = (h16*)alloc((size_t)RD * 2);
  h16* vbuf = (h16*)alloc((size_t)RD * 2);
  h16* qb = (h16*)alloc((size_t)RD * 2);
  h16* ka = (h16*)alloc((size_t)RD * 2);
  h16* va = (h16*)alloc((size_t)RD * 2);
  h16* vt1 = (h16*)alloc((size_t)kB * kNH * kHD * kVTLD * 2);
  h16* vt2 = (h16*)alloc((size_t)kB * kNH * kHD * kVTLD * 2);
  h16* concat16 = (h16*)alloc((size_t)kR * 2 * kD * 2);
  float* fused32 = (float*)alloc((size_t)RD * 4);
  float* xa1_32 = (float*)alloc((size_t)RD * 4);
  h16* xa1_16 = (h16*)alloc((size_t)RD * 2);
  h16* h1_16 = (h16*)alloc((size_t)kR * 4 * kD * 2);
  float* ffn32 = (float*)alloc((size_t)RD * 4);
  float* tmpA = (float*)alloc((size_t)RD * 4);
  float* tmpB = (float*)alloc((size_t)RD * 4);

  const int WSZ = kD * kD;
  auto run_block = [&](int i, int xa, int xb, float* outp) {
    gemm(s16[xa], kD, wq_a16 + i * WSZ, bq_a + i * kD, nullptr, qa, kD, kD, kD, 0);
    gemm(s16[xb], kD, wk_b16 + i * WSZ, bk_b + i * kD, nullptr, kbuf, kD, kD, kD, 0);
    gemm(s16[xb], kD, wv_b16 + i * WSZ, bv_b + i * kD, nullptr, vbuf, kD, kD, kD, 0);
    gemm(s16[xb], kD, wq_b16 + i * WSZ, bq_b + i * kD, nullptr, qb, kD, kD, kD, 0);
    gemm(s16[xa], kD, wk_a16 + i * WSZ, bk_a + i * kD, nullptr, ka, kD, kD, kD, 0);
    gemm(s16[xa], kD, wv_a16 + i * WSZ, bv_a + i * kD, nullptr, va, kD, kD, kD, 0);
    vtrans(vbuf, kD, vt1);
    vtrans(va, kD, vt2);
    attn(qa, kD, kbuf, kD, vt1, concat16, 2 * kD);          // a2b -> cols [0,256)
    attn(qb, kD, ka, kD, vt2, concat16 + kD, 2 * kD);       // b2a -> cols [256,512)
    gemm(concat16, 2 * kD, wfuse16 + i * 2 * WSZ, bfuse + i * kD,
         fused32, nullptr, kD, 2 * kD, kD, 0);
    ln(s32[xa], fused32, gamma + i, ln1_g + i * kD, ln1_b + i * kD,
       xa1_32, xa1_16, kD);
    gemm(xa1_16, kD, w1_16 + i * 4 * WSZ, b1 + i * 4 * kD,
         nullptr, h1_16, 4 * kD, kD, 4 * kD, 1);            // fused GELU
    gemm(h1_16, 4 * kD, w2_16 + i * 4 * WSZ, b2 + i * kD,
         ffn32, nullptr, kD, 4 * kD, kD, 0);
    ln(xa1_32, ffn32, gamma_ffn + i, ln2_g + i * kD, ln2_b + i * kD,
       outp, nullptr, kD);
  };
  auto avg = [&](float* dst) {
    k_avg<<<(RD + 255) / 256, 256, 0, stream>>>(tmpA, tmpB, dst, RD);
  };

  run_block(0, 0, 1, tmpA); run_block(1, 0, 3, tmpB); avg(e32[0]);
  run_block(0, 1, 0, tmpA); run_block(2, 1, 2, tmpB); avg(e32[1]);
  run_block(2, 2, 1, tmpA); run_block(3, 2, 3, tmpB); avg(e32[2]);
  run_block(1, 3, 0, tmpA); run_block(3, 3, 2, tmpB); avg(e32[3]);

  // ---- head-phase transient arena (reuse block scratch)
  p = transBase;
  h16* lnE16 = (h16*)alloc((size_t)RD * 2);
  float* allf32 = (float*)alloc((size_t)kR * 4 * kD * 4);
  h16* allfln16 = (h16*)alloc((size_t)kR * 4 * kD * 2);
  h16* x16 = (h16*)alloc((size_t)RD * 2);
  h16* qkv16 = (h16*)alloc((size_t)kR * 3 * kD * 2);
  h16* vt3 = (h16*)alloc((size_t)kB * kNH * kHD * kVTLD * 2);
  h16* attn16 = (h16*)alloc((size_t)RD * 2);
  float* ao32 = (float*)alloc((size_t)RD * 4);
  float* xfin32 = (float*)alloc((size_t)RD * 4);

  for (int i = 0; i < 4; ++i) {
    ln(e32[i], nullptr, nullptr, fp_ln_g, fp_ln_b, nullptr, lnE16, kD);
    gemm(lnE16, kD, fpW16, fp_b, allf32 + i * kD, nullptr, 4 * kD, kD, kD, 1);
  }
  ln(allf32, nullptr, nullptr, ff_ln_g, ff_ln_b, nullptr, allfln16, 4 * kD);
  gemm(allfln16, 4 * kD, ffW16, ff_b, nullptr, x16, kD, 4 * kD, kD, 1);
  gemm(x16, kD, wqkv16, mha_bqkv, nullptr, qkv16, 3 * kD, kD, 3 * kD, 0);
  vtrans(qkv16 + 2 * kD, 3 * kD, vt3);
  attn(qkv16, 3 * kD, qkv16 + kD, 3 * kD, vt3, attn16, kD);
  gemm(attn16, kD, wo16, mha_bo, ao32, nullptr, kD, kD, kD, 0);
  ln(ao32, nullptr, nullptr, norm_g, norm_b, xfin32, nullptr, kD);
  k_from_seq<<<RD / 256, 256, 0, stream>>>(xfin32, (float*)d_out);
}
